// EncoderLayer_17179869813
// MI455X (gfx1250) — matrix operand order repaired and checked by reading, not compiled, not run on hardware
//
#include <hip/hip_runtime.h>
#include <hip/hip_bf16.h>

typedef __attribute__((ext_vector_type(16))) _Float16 v16h;
typedef __attribute__((ext_vector_type(8)))  _Float16 v8h;
typedef __attribute__((ext_vector_type(8)))  float    v8f;
typedef __attribute__((ext_vector_type(4)))  unsigned u32x4;
typedef __attribute__((ext_vector_type(8)))  int      i32x8;
typedef __attribute__((ext_vector_type(4)))  int      i32x4;
typedef _Float16 half_t;

#define WMMA_F16(a, b, c) \
  __builtin_amdgcn_wmma_f32_16x16x32_f16(false, (a), false, (b), (short)0, (c), false, false)

static constexpr int EMBED = 512;
static constexpr int HEADS = 8;
static constexpr int HEAD_DIM = 64;
static constexpr int FF = 2048;
static constexpr int BATCH = 4;
static constexpr int SEQ = 2048;
static constexpr int MTOT = BATCH * SEQ;  // 8192

// ---------------------------------------------------------------------------
// TDM: issue a 2D f16 tile DMA (global -> LDS). Descriptor layout per CDNA5
// ISA §8.3/8.4. tile_dim0 elements along contiguous dim0 (stride 1), tile_dim1
// rows with row stride `stride_elems`; data lands packed row-major in LDS.
// Tracked with TENSORcnt. All operands are uniform -> SGPR groups.
// This toolchain exposes the 6-arg builtin (clang-23 / therock-10.0 form).
// ---------------------------------------------------------------------------
__device__ __forceinline__ void tdm_load_tile_2d(unsigned lds_off, const half_t* gptr,
                                                 int dim0, int dim1, int tile0, int tile1,
                                                 int stride_elems) {
  unsigned long long ga = (unsigned long long)(size_t)(const void*)gptr;
  u32x4 g0;
  g0[0] = 1u;                                              // count=1, user descriptor
  g0[1] = lds_off;                                         // LDS byte address
  g0[2] = (unsigned)ga;                                    // global addr [31:0]
  g0[3] = (unsigned)((ga >> 32) & 0x1FFFFFFull) | (2u << 30);  // addr[56:32] | type=2
  i32x8 g1;
  g1[0] = (int)(1u << 16);                                 // data_size=1 (2 bytes)
  g1[1] = (int)(((unsigned)dim0 & 0xFFFFu) << 16);         // tensor_dim0 [79:48]
  g1[2] = (int)(((unsigned)dim0 >> 16) | (((unsigned)dim1 & 0xFFFFu) << 16));
  g1[3] = (int)(((unsigned)dim1 >> 16) | ((unsigned)tile0 << 16));  // tile_dim0 [127:112]
  g1[4] = (int)((unsigned)tile1 & 0xFFFFu);                // tile_dim1 [143:128], tile_dim2=0
  g1[5] = stride_elems;                                    // tensor_dim0_stride low 32
  g1[6] = 0;                                               // stride hi16, dim1_stride lo16
  g1[7] = 0;
  i32x4 z4;
  z4[0] = 0; z4[1] = 0; z4[2] = 0; z4[3] = 0;
  i32x8 z8;
#pragma unroll
  for (int i = 0; i < 8; ++i) z8[i] = 0;
  __builtin_amdgcn_tensor_load_to_lds(g0, g1, z4, z4, z8, 0);
}

// ---------------------------------------------------------------------------
// fp32 -> f16 conversion (n must be a multiple of 4; all our sizes are)
// ---------------------------------------------------------------------------
__global__ void cvt_f32_f16(const float* __restrict__ src, half_t* __restrict__ dst, int n) {
  int i = (blockIdx.x * blockDim.x + threadIdx.x) * 4;
  if (i + 3 < n) {
    float4 f = *reinterpret_cast<const float4*>(src + i);
    dst[i + 0] = (half_t)f.x;
    dst[i + 1] = (half_t)f.y;
    dst[i + 2] = (half_t)f.z;
    dst[i + 3] = (half_t)f.w;
  }
}

// ---------------------------------------------------------------------------
// fp32 [K][N] -> f16 [N][K] transposed conversion (one-time weight prep so the
// GEMM never has to transpose in its hot loop). Tiled 32x32 through LDS.
// ---------------------------------------------------------------------------
__global__ __launch_bounds__(256) void cvt_transpose_f32_f16(
    const float* __restrict__ src, half_t* __restrict__ dst, int K, int N) {
  __shared__ float tile[32][33];
  const int k0 = blockIdx.y * 32;
  const int n0 = blockIdx.x * 32;
  const int tx = threadIdx.x & 31;
  const int ty = threadIdx.x >> 5;  // 0..7
#pragma unroll
  for (int i = 0; i < 32; i += 8)
    tile[ty + i][tx] = src[(size_t)(k0 + ty + i) * N + n0 + tx];
  __syncthreads();
#pragma unroll
  for (int i = 0; i < 32; i += 8)
    dst[(size_t)(n0 + ty + i) * K + k0 + tx] = (half_t)tile[tx][ty + i];
}

// ---------------------------------------------------------------------------
// WMMA GEMM: C[M,N] = A[M,K] * B[K,N] + bias, B supplied pre-transposed as
// Bt[N][K] f16. Block = 256 threads = 8 waves (2x4); tile 64(M) x 128(N),
// K-step 32. Tiles are DMA'd into double-buffered LDS by the Tensor Data
// Mover (wave 0 issues, TENSORcnt + barrier publishes), so all 8 waves stay
// on the WMMA pipe.
// EPI 0: out32[m*N+n] = acc + bias
// EPI 1: out16[m*N+n] = max(acc+bias, 0)           (ReLU, f16)
// EPI 2: head-split f16 (Q/K): out16[((b*H+h)*S+s)*64+d] = (acc+bias)*scale
// EPI 3: head-split transposed f16 (V): out16[((b*H+h)*64+d)*S+s] = acc+bias
// ---------------------------------------------------------------------------
template <int EPI>
__global__ __launch_bounds__(256) void gemm_wmma(
    const half_t* __restrict__ A, const half_t* __restrict__ Bt,
    const float* __restrict__ bias, float* __restrict__ out32,
    half_t* __restrict__ out16, int M, int N, int K, float scale) {
  __shared__ half_t As[2][64 * 32];    // [m][k], double buffered
  __shared__ half_t Bs[2][128 * 32];   // [n][k], double buffered

  const int tid = threadIdx.x;
  const int lane = tid & 31;
  const int wid = tid >> 5;
  const int wrow = wid >> 2;   // 0..1
  const int wcol = wid & 3;    // 0..3
  const int bm = blockIdx.y * 64;
  const int bn = blockIdx.x * 128;

  const int l15 = lane & 15;
  const int kA = (lane < 16) ? 0 : 8;    // A-frag per-lane K base
  const int kB = (lane < 16) ? 0 : 8;    // B-frag per-lane K base

  const unsigned offA[2] = {(unsigned)(size_t)(void*)&As[0][0],
                            (unsigned)(size_t)(void*)&As[1][0]};
  const unsigned offB[2] = {(unsigned)(size_t)(void*)&Bs[0][0],
                            (unsigned)(size_t)(void*)&Bs[1][0]};

  v8f acc[2][2];
#pragma unroll
  for (int i = 0; i < 2; ++i)
#pragma unroll
    for (int j = 0; j < 2; ++j)
#pragma unroll
      for (int e = 0; e < 8; ++e) acc[i][j][e] = 0.0f;

  // prologue: DMA first K tiles into buffer 0
  if (wid == 0) {
    tdm_load_tile_2d(offA[0], A + (size_t)bm * K, K, M, 32, 64, K);
    tdm_load_tile_2d(offB[0], Bt + (size_t)bn * K, K, N, 32, 128, K);
  }

  int buf = 0;
  for (int k0 = 0; k0 < K; k0 += 32) {
    if (wid == 0) {
      if (k0 + 32 < K) {
        // issue next tiles into the other buffer, then wait for current pair
        tdm_load_tile_2d(offA[buf ^ 1], A + (size_t)bm * K + k0 + 32, K, M, 32, 64, K);
        tdm_load_tile_2d(offB[buf ^ 1], Bt + (size_t)bn * K + k0 + 32, K, N, 32, 128, K);
        __builtin_amdgcn_s_wait_tensorcnt(2);
      } else {
        __builtin_amdgcn_s_wait_tensorcnt(0);
      }
    }
    __syncthreads();  // publish DMA'd tiles to all waves

    const half_t* Ab = &As[buf][0];
    const half_t* Bb = &Bs[buf][0];
    v16h afr[2], bfr[2];
#pragma unroll
    for (int i = 0; i < 2; ++i) {
      int mrow = wrow * 32 + i * 16 + l15;
      v8h lo = *reinterpret_cast<const v8h*>(&Ab[mrow * 32 + kA]);
      v8h hi = *reinterpret_cast<const v8h*>(&Ab[mrow * 32 + 16 + kA]);
#pragma unroll
      for (int e = 0; e < 8; ++e) { afr[i][e] = lo[e]; afr[i][e + 8] = hi[e]; }
    }
#pragma unroll
    for (int j = 0; j < 2; ++j) {
      int ncol = wcol * 32 + j * 16 + l15;
      v8h lo = *reinterpret_cast<const v8h*>(&Bb[ncol * 32 + kB]);
      v8h hi = *reinterpret_cast<const v8h*>(&Bb[ncol * 32 + kB + 16]);
#pragma unroll
      for (int e = 0; e < 8; ++e) { bfr[j][e] = lo[e]; bfr[j][e + 8] = hi[e]; }
    }
#pragma unroll
    for (int i = 0; i < 2; ++i)
#pragma unroll
      for (int j = 0; j < 2; ++j) acc[i][j] = WMMA_F16(afr[i], bfr[j], acc[i][j]);
    __syncthreads();  // all waves done reading before this buffer is overwritten
    buf ^= 1;
  }

  // epilogue: C layout — VGPR v: lanes 0-15 (M=v, N=lane), lanes 16-31 (M=v+8, N=lane-16)
  const int rowoff = (lane < 16) ? 0 : 8;
#pragma unroll
  for (int i = 0; i < 2; ++i) {
#pragma unroll
    for (int j = 0; j < 2; ++j) {
      int n = bn + wcol * 32 + j * 16 + l15;
      float bv = bias[n];
#pragma unroll
      for (int v = 0; v < 8; ++v) {
        int m = bm + wrow * 32 + i * 16 + v + rowoff;
        float val = acc[i][j][v] + bv;
        if (EPI == 0) {
          out32[(size_t)m * N + n] = val;
        } else if (EPI == 1) {
          out16[(size_t)m * N + n] = (half_t)fmaxf(val, 0.0f);
        } else if (EPI == 2) {  // [B,H,S,Dh]
          int b = m / SEQ, s = m % SEQ;
          int h = n >> 6, d = n & 63;
          out16[(((size_t)(b * HEADS + h)) * SEQ + s) * HEAD_DIM + d] = (half_t)(val * scale);
        } else {                // [B,H,Dh,S]
          int b = m / SEQ, s = m % SEQ;
          int h = n >> 6, d = n & 63;
          out16[(((size_t)(b * HEADS + h)) * HEAD_DIM + d) * SEQ + s] = (half_t)val;
        }
      }
    }
  }
}

// ---------------------------------------------------------------------------
// Flash attention. Q,K f16 [B,H,S,64]; V f16 transposed [B,H,64,S];
// mask int32 [B,S] over keys. Block = 128 threads = 4 waves; each wave owns a
// 16-row Q tile; 32 keys/iter. Output: context f16 [B,S,E]. Softmax scale is
// pre-folded into Q.
// ---------------------------------------------------------------------------
__global__ __launch_bounds__(128) void attn_kernel(
    const half_t* __restrict__ Q, const half_t* __restrict__ Kk,
    const half_t* __restrict__ Vt, const int* __restrict__ mask,
    half_t* __restrict__ ctx) {
  __shared__ half_t Ks[32 * 64];     // [key][d]
  __shared__ half_t Vs[64 * 32];     // [d][key]
  __shared__ half_t Ps[4][16 * 32];  // per-wave P transpose scratch [m][k]

  const int tid = threadIdx.x;
  const int lane = tid & 31;
  const int wid = tid >> 5;

  const int nQt = SEQ / 64;                 // 32
  const int bh = blockIdx.x / nQt;          // 0..B*H-1
  const int qt = blockIdx.x % nQt;
  const int b = bh / HEADS;
  const int h = bh % HEADS;

  const half_t* Qh = Q + (size_t)bh * SEQ * HEAD_DIM;
  const half_t* Kh = Kk + (size_t)bh * SEQ * HEAD_DIM;
  const half_t* Vth = Vt + (size_t)bh * HEAD_DIM * SEQ;
  const int* maskb = mask + b * SEQ;

  const int qs = qt * 64 + wid * 16;        // first query row of this wave
  const int l15 = lane & 15;
  const int kA = (lane < 16) ? 0 : 8;
  const int kB = (lane < 16) ? 0 : 8;

  // load Q A-fragments once: q0 covers d=0..31, q1 covers d=32..63
  v16h q0, q1;
  {
    const half_t* qrow = Qh + (size_t)(qs + l15) * HEAD_DIM;
    v8h a0 = *reinterpret_cast<const v8h*>(qrow + kA);
    v8h a1 = *reinterpret_cast<const v8h*>(qrow + 16 + kA);
    v8h a2 = *reinterpret_cast<const v8h*>(qrow + 32 + kA);
    v8h a3 = *reinterpret_cast<const v8h*>(qrow + 48 + kA);
#pragma unroll
    for (int e = 0; e < 8; ++e) {
      q0[e] = a0[e]; q0[e + 8] = a1[e];
      q1[e] = a2[e]; q1[e + 8] = a3[e];
    }
  }

  float mrow[8], lrow[8];
  v8f O[4];
#pragma unroll
  for (int v = 0; v < 8; ++v) { mrow[v] = -1e30f; lrow[v] = 0.0f; }
#pragma unroll
  for (int nt = 0; nt < 4; ++nt)
#pragma unroll
    for (int e = 0; e < 8; ++e) O[nt][e] = 0.0f;

  for (int kc = 0; kc < SEQ; kc += 32) {
    // cooperative stage of K [key][d] and V [d][key] — all vectorized copies
#pragma unroll
    for (int it = 0; it < 2; ++it) {
      int ch = tid + it * 128;       // 256 chunks of 8 halfs each
      int kr = ch >> 3;              // key row 0..31
      int kf = (ch & 7) * 8;         // d offset
      v8h kv = *reinterpret_cast<const v8h*>(Kh + (size_t)(kc + kr) * HEAD_DIM + kf);
      *reinterpret_cast<v8h*>(&Ks[kr * 64 + kf]) = kv;
      int dr = ch >> 2;              // d row 0..63
      int so = (ch & 3) * 8;         // key offset 0..24
      v8h vv = *reinterpret_cast<const v8h*>(Vth + (size_t)dr * SEQ + kc + so);
      *reinterpret_cast<v8h*>(&Vs[dr * 32 + so]) = vv;
    }
    __syncthreads();

    // scores: sc[j] = Q(16x64) x K^T for keys [kc + j*16, +16)
    v8f sc[2];
#pragma unroll
    for (int j = 0; j < 2; ++j) {
      int krow = j * 16 + l15;       // this lane's key column (local)
      v16h b0, b1;
      v8h x0 = *reinterpret_cast<const v8h*>(&Ks[krow * 64 + kB]);
      v8h x1 = *reinterpret_cast<const v8h*>(&Ks[krow * 64 + kB + 16]);
      v8h x2 = *reinterpret_cast<const v8h*>(&Ks[krow * 64 + 32 + kB]);
      v8h x3 = *reinterpret_cast<const v8h*>(&Ks[krow * 64 + 32 + kB + 16]);
#pragma unroll
      for (int e = 0; e < 8; ++e) {
        b0[e] = x0[e]; b0[e + 8] = x1[e];
        b1[e] = x2[e]; b1[e + 8] = x3[e];
      }
      v8f z;
#pragma unroll
      for (int e = 0; e < 8; ++e) z[e] = 0.0f;
      z = WMMA_F16(q0, b0, z);
      z = WMMA_F16(q1, b1, z);
      sc[j] = z;
    }

    // mask (per key column)
    int mk0 = maskb[kc + l15];
    int mk1 = maskb[kc + 16 + l15];
#pragma unroll
    for (int v = 0; v < 8; ++v) {
      if (mk0 == 0) sc[0][v] = -1e20f;
      if (mk1 == 0) sc[1][v] = -1e20f;
    }

    // online softmax; row reductions across the 16 lanes holding each row
    float p0[8], p1[8];
#pragma unroll
    for (int v = 0; v < 8; ++v) {
      float t = fmaxf(sc[0][v], sc[1][v]);
#pragma unroll
      for (int sh = 1; sh < 16; sh <<= 1) t = fmaxf(t, __shfl_xor(t, sh, 32));
      float mnew = fmaxf(mrow[v], t);
      float alpha = __expf(mrow[v] - mnew);
      mrow[v] = mnew;
      p0[v] = __expf(sc[0][v] - mnew);
      p1[v] = __expf(sc[1][v] - mnew);
      float rs = p0[v] + p1[v];
#pragma unroll
      for (int sh = 1; sh < 16; sh <<= 1) rs += __shfl_xor(rs, sh, 32);
      lrow[v] = lrow[v] * alpha + rs;
#pragma unroll
      for (int nt = 0; nt < 4; ++nt) O[nt][v] *= alpha;
    }

    // transpose P (C-layout -> A-layout) via per-wave LDS scratch
    half_t* Pw = &Ps[wid][0];
    const int prow = (lane < 16) ? 0 : 8;
#pragma unroll
    for (int v = 0; v < 8; ++v) {
      Pw[(v + prow) * 32 + l15] = (half_t)p0[v];
      Pw[(v + prow) * 32 + 16 + l15] = (half_t)p1[v];
    }
    asm volatile("s_wait_dscnt 0" ::: "memory");  // same-wave LDS store->load
    v16h pa;
    {
      v8h lo = *reinterpret_cast<const v8h*>(&Pw[l15 * 32 + kA]);
      v8h hi = *reinterpret_cast<const v8h*>(&Pw[l15 * 32 + 16 + kA]);
#pragma unroll
      for (int e = 0; e < 8; ++e) { pa[e] = lo[e]; pa[e + 8] = hi[e]; }
    }

    // O += P(16x32) x V(32x64)
#pragma unroll
    for (int nt = 0; nt < 4; ++nt) {
      int dcol = nt * 16 + l15;
      v16h bvv;
      v8h lo = *reinterpret_cast<const v8h*>(&Vs[dcol * 32 + kB]);
      v8h hi = *reinterpret_cast<const v8h*>(&Vs[dcol * 32 + kB + 16]);
#pragma unroll
      for (int e = 0; e < 8; ++e) { bvv[e] = lo[e]; bvv[e + 8] = hi[e]; }
      O[nt] = WMMA_F16(pa, bvv, O[nt]);
    }
    __syncthreads();
  }

  // normalize and store context f16 as [B,S,E]
  const int rowoff = (lane < 16) ? 0 : 8;
#pragma unroll
  for (int v = 0; v < 8; ++v) {
    float inv = 1.0f / lrow[v];
    int s = qs + v + rowoff;
    size_t base = ((size_t)(b * SEQ + s)) * EMBED + h * HEAD_DIM;
#pragma unroll
    for (int nt = 0; nt < 4; ++nt)
      ctx[base + nt * 16 + l15] = (half_t)(O[nt][v] * inv);
  }
}

// ---------------------------------------------------------------------------
// LayerNorm( a + r ) * g + beta ; one wave per 512-wide row, 4 rows per block.
// Always writes out32; optionally also writes an f16 copy.
// ---------------------------------------------------------------------------
template <bool W16>
__global__ __launch_bounds__(128) void ln_residual(
    const float* __restrict__ a, const float* __restrict__ r,
    const float* __restrict__ g, const float* __restrict__ beta,
    float* __restrict__ out32, half_t* __restrict__ out16) {
  const int lane = threadIdx.x & 31;
  const int wid = threadIdx.x >> 5;
  const int row = blockIdx.x * 4 + wid;
  const float* ar = a + (size_t)row * EMBED;
  const float* rr = r + (size_t)row * EMBED;

  float vals[16];
  float sum = 0.0f;
#pragma unroll
  for (int i = 0; i < 16; ++i) {
    int c = lane + i * 32;
    vals[i] = ar[c] + rr[c];
    sum += vals[i];
  }
#pragma unroll
  for (int sh = 1; sh < 32; sh <<= 1) sum += __shfl_xor(sum, sh, 32);
  float mu = sum * (1.0f / (float)EMBED);
  float vs = 0.0f;
#pragma unroll
  for (int i = 0; i < 16; ++i) {
    float d = vals[i] - mu;
    vs += d * d;
  }
#pragma unroll
  for (int sh = 1; sh < 32; sh <<= 1) vs += __shfl_xor(vs, sh, 32);
  float rstd = rsqrtf(vs * (1.0f / (float)EMBED) + 1e-5f);
#pragma unroll
  for (int i = 0; i < 16; ++i) {
    int c = lane + i * 32;
    float o = (vals[i] - mu) * rstd * g[c] + beta[c];
    out32[(size_t)row * EMBED + c] = o;
    if (W16) out16[(size_t)row * EMBED + c] = (half_t)o;
  }
}

// ---------------------------------------------------------------------------
extern "C" void kernel_launch(void* const* d_in, const int* in_sizes, int n_in,
                              void* d_out, int out_size, void* d_ws, size_t ws_size,
                              hipStream_t stream) {
  (void)in_sizes; (void)n_in; (void)out_size; (void)ws_size;

  const float* x   = (const float*)d_in[0];
  const int*   msk = (const int*)d_in[1];
  const float* Wq  = (const float*)d_in[2];
  const float* bq  = (const float*)d_in[3];
  const float* Wk  = (const float*)d_in[4];
  const float* bk  = (const float*)d_in[5];
  const float* Wv  = (const float*)d_in[6];
  const float* bv  = (const float*)d_in[7];
  const float* Wo  = (const float*)d_in[8];
  const float* bo  = (const float*)d_in[9];
  const float* g1  = (const float*)d_in[10];
  const float* b1  = (const float*)d_in[11];
  const float* g2  = (const float*)d_in[12];
  const float* b2  = (const float*)d_in[13];
  const float* W1  = (const float*)d_in[14];
  const float* b1f = (const float*)d_in[15];
  const float* W2  = (const float*)d_in[16];
  const float* b2f = (const float*)d_in[17];
  float* out = (float*)d_out;

  size_t off = 0;
  auto take = [&](size_t bytes) -> void* {
    void* p = (char*)d_ws + off;
    off += (bytes + 255) & ~(size_t)255;
    return p;
  };
  half_t* x16   = (half_t*)take((size_t)MTOT * EMBED * 2);
  half_t* WqT   = (half_t*)take((size_t)EMBED * EMBED * 2);  // [N][K] transposed f16
  half_t* WkT   = (half_t*)take((size_t)EMBED * EMBED * 2);
  half_t* WvT   = (half_t*)take((size_t)EMBED * EMBED * 2);
  half_t* WoT   = (half_t*)take((size_t)EMBED * EMBED * 2);
  half_t* W1T   = (half_t*)take((size_t)EMBED * FF * 2);     // [FF][EMBED]
  half_t* W2T   = (half_t*)take((size_t)FF * EMBED * 2);     // [EMBED][FF]
  half_t* Q16   = (half_t*)take((size_t)MTOT * EMBED * 2);   // [B,H,S,Dh]
  half_t* K16   = (half_t*)take((size_t)MTOT * EMBED * 2);   // [B,H,S,Dh]
  half_t* V16t  = (half_t*)take((size_t)MTOT * EMBED * 2);   // [B,H,Dh,S]
  half_t* ctx16 = (half_t*)take((size_t)MTOT * EMBED * 2);
  float*  attnp = (float*) take((size_t)MTOT * EMBED * 4);
  float*  h32   = (float*) take((size_t)MTOT * EMBED * 4);
  half_t* h16   = (half_t*)take((size_t)MTOT * EMBED * 2);
  half_t* ff1   = (half_t*)take((size_t)MTOT * FF * 2);
  float*  ff2   = (float*) take((size_t)MTOT * EMBED * 4);

  // activations: straight f32->f16
  {
    int n = MTOT * EMBED;
    cvt_f32_f16<<<(n / 4 + 255) / 256, 256, 0, stream>>>(x, x16, n);
  }
  // weights: transpose during conversion ([K][N] f32 -> [N][K] f16)
  auto cvtT = [&](const float* s, half_t* d, int K, int N) {
    dim3 g(N / 32, K / 32);
    cvt_transpose_f32_f16<<<g, 256, 0, stream>>>(s, d, K, N);
  };
  cvtT(Wq, WqT, EMBED, EMBED);
  cvtT(Wk, WkT, EMBED, EMBED);
  cvtT(Wv, WvT, EMBED, EMBED);
  cvtT(Wo, WoT, EMBED, EMBED);
  cvtT(W1, W1T, EMBED, FF);
  cvtT(W2, W2T, FF, EMBED);

  dim3 gE(EMBED / 128, MTOT / 64);   // (4, 128)
  dim3 gF(FF / 128,    MTOT / 64);   // (16, 128)

  // QKV projections (1/sqrt(64) folded into Q; V stored head-transposed)
  gemm_wmma<2><<<gE, 256, 0, stream>>>(x16, WqT, bq, nullptr, Q16, MTOT, EMBED, EMBED, 0.125f);
  gemm_wmma<2><<<gE, 256, 0, stream>>>(x16, WkT, bk, nullptr, K16, MTOT, EMBED, EMBED, 1.0f);
  gemm_wmma<3><<<gE, 256, 0, stream>>>(x16, WvT, bv, nullptr, V16t, MTOT, EMBED, EMBED, 1.0f);

  // attention
  attn_kernel<<<BATCH * HEADS * (SEQ / 64), 128, 0, stream>>>(Q16, K16, V16t, msk, ctx16);

  // output projection
  gemm_wmma<0><<<gE, 256, 0, stream>>>(ctx16, WoT, bo, attnp, nullptr, MTOT, EMBED, EMBED, 1.0f);

  // h = LN(x + attn_out)
  ln_residual<true><<<MTOT / 4, 128, 0, stream>>>(attnp, x, g1, b1, h32, h16);

  // FFN
  gemm_wmma<1><<<gF, 256, 0, stream>>>(h16, W1T, b1f, nullptr, ff1, MTOT, FF, EMBED, 1.0f);
  gemm_wmma<0><<<gE, 256, 0, stream>>>(ff1, W2T, b2f, ff2, nullptr, MTOT, EMBED, FF, 1.0f);

  // out = LN(h + ff)
  ln_residual<false><<<MTOT / 4, 128, 0, stream>>>(ff2, h32, g2, b2, out, nullptr);
}
